// TokenGNNEncoderLayer_22737556865605
// MI455X (gfx1250) — matrix operand order repaired and checked by reading, not compiled, not run on hardware
//
#include <hip/hip_runtime.h>
#include <hip/hip_bf16.h>

#define NNODES 50000
#define NEDGES 800000
#define DMODEL 128
#define NHEADS 8
#define NFF    512
#define NTYPES 5

typedef __attribute__((ext_vector_type(16))) __bf16 bf16x16;
typedef __attribute__((ext_vector_type(8)))  __bf16 bf16x8;
typedef __attribute__((ext_vector_type(8)))  float  f32x8;

__device__ __forceinline__ float bf2f(__bf16 v) {
  unsigned short s = __builtin_bit_cast(unsigned short, v);
  unsigned u = ((unsigned)s) << 16;
  return __builtin_bit_cast(float, u);
}
__device__ __forceinline__ __bf16 f2bf(float f) {
  unsigned u = __builtin_bit_cast(unsigned, f);
  u += 0x7FFFu + ((u >> 16) & 1u);      // round to nearest even
  unsigned short s = (unsigned short)(u >> 16);
  return __builtin_bit_cast(__bf16, s);
}

// exact IEEE float atomic max via signed/unsigned int ordering trick
__device__ __forceinline__ void atomicMaxFloat(float* addr, float val) {
  if (val >= 0.0f) atomicMax((int*)addr, __float_as_int(val));
  else             atomicMin((unsigned int*)addr, (unsigned int)__float_as_int(val));
}

__device__ __forceinline__ f32x8 wmma_bf16(bf16x16 a, bf16x16 b, f32x8 c) {
  return __builtin_amdgcn_wmma_f32_16x16x32_bf16(false, a, false, b, (short)0, c, false, false);
}

// ---------------------------------------------------------------------------
// Async-stage a contiguous global span into LDS via the CDNA5 async LDS pipe.
// Each issued instruction moves 16B/lane (b128); tracked by ASYNCcnt.
// Caller must __syncthreads() afterwards (wait covers only this wave's issues).
// ---------------------------------------------------------------------------
__device__ __forceinline__ void async_copy_tile(const void* gsrc, void* lds,
                                                int bytes, int tid, int nthreads) {
  unsigned           lbase = (unsigned)(uintptr_t)lds;
  unsigned long long gbase = (unsigned long long)(uintptr_t)gsrc;
#pragma unroll
  for (int off = 0; off < bytes; off += nthreads * 16) {
    unsigned           l = lbase + off + tid * 16;
    unsigned long long g = gbase + (unsigned long long)(off + tid * 16);
    asm volatile("global_load_async_to_lds_b128 %0, %1, off" :: "v"(l), "v"(g) : "memory");
  }
  asm volatile("s_wait_asynccnt 0x0" ::: "memory");
}

// ---------------------------------------------------------------------------
// One wave computes a 16x64 strip: 4 accumulators share one A fragment.
// A read from LDS (row-major [16,K] strip), B from transposed weights BT
// (row-major [cols, K], ldb == K).
// A frag per ISA: lane<16 -> row=lane,    K = k0+{0..7, 16..23};
//                 lane>=16 -> row=lane-16, K = k0+{8..15, 24..31}.
// B frag: lane column = lane&15, same K order as A.
// ---------------------------------------------------------------------------
template<int K>
__device__ __forceinline__ void wmma_row4(const __bf16* Alds, const __bf16* BT,
                                          int lane, f32x8 (&acc)[4]) {
  const int row = lane & 15;
  const int khi = lane >> 4;
  const __bf16* arow = Alds + row * K;
  const __bf16* brow = BT + (size_t)(lane & 15) * K;
#pragma unroll
  for (int k0 = 0; k0 < K; k0 += 32) {
    bf16x8 lo = *(const bf16x8*)(arow + k0 + khi * 8);
    bf16x8 hi = *(const bf16x8*)(arow + k0 + 16 + khi * 8);
    bf16x16 a = __builtin_shufflevector(lo, hi, 0,1,2,3,4,5,6,7,8,9,10,11,12,13,14,15);
#pragma unroll
    for (int i = 0; i < 4; ++i) {
      bf16x8 blo = *(const bf16x8*)(brow + (size_t)i * 16 * K + k0 + khi * 8);
      bf16x8 bhi = *(const bf16x8*)(brow + (size_t)i * 16 * K + k0 + 16 + khi * 8);
      bf16x16 b = __builtin_shufflevector(blo, bhi, 0,1,2,3,4,5,6,7,8,9,10,11,12,13,14,15);
      acc[i] = wmma_bf16(a, b, acc[i]);
    }
  }
}

// ---------------------------------------------------------------------------
// Weight prep: transpose + convert to bf16
// ---------------------------------------------------------------------------
__global__ void transpose_wk_kernel(const float* __restrict__ Wk, __bf16* __restrict__ WkT) {
  int idx = blockIdx.x * blockDim.x + threadIdx.x;
  if (idx >= NTYPES * DMODEL * DMODEL) return;
  int t = idx / (DMODEL * DMODEL);
  int r = idx % (DMODEL * DMODEL);
  int n = r / DMODEL, k = r % DMODEL;
  WkT[idx] = f2bf(Wk[t * DMODEL * DMODEL + k * DMODEL + n]);
}
__global__ void transpose_w1_kernel(const float* __restrict__ W1, __bf16* __restrict__ W1T) {
  int idx = blockIdx.x * blockDim.x + threadIdx.x;       // W1T [512,128]
  if (idx >= NFF * DMODEL) return;
  int n = idx / DMODEL, k = idx % DMODEL;
  W1T[idx] = f2bf(W1[k * NFF + n]);
}
__global__ void transpose_w2_kernel(const float* __restrict__ W2, __bf16* __restrict__ W2T) {
  int idx = blockIdx.x * blockDim.x + threadIdx.x;       // W2T [128,512]
  if (idx >= DMODEL * NFF) return;
  int n = idx / NFF, k = idx % NFF;
  W2T[idx] = f2bf(W2[k * DMODEL + n]);
}

__global__ void etype_kernel(const unsigned char* __restrict__ masks,
                             unsigned char* __restrict__ et) {
  int e = blockIdx.x * blockDim.x + threadIdx.x;
  if (e >= NEDGES) return;
  unsigned char t = 0;
#pragma unroll
  for (int i = 1; i < NTYPES; ++i)
    if (masks[(size_t)i * NEDGES + e]) t = (unsigned char)i;
  et[e] = t;
}

__global__ void init_kernel(float* __restrict__ m, float* __restrict__ den,
                            float* __restrict__ attn) {
  int i = blockIdx.x * blockDim.x + threadIdx.x;
  if (i < NNODES * NHEADS) { m[i] = -__builtin_inff(); den[i] = 0.0f; }
  if (i < NNODES * DMODEL) attn[i] = 0.0f;
}

// ---------------------------------------------------------------------------
// LayerNorm 1: z = LN(x) -> bf16
// ---------------------------------------------------------------------------
__global__ void ln1_kernel(const float* __restrict__ x, const float* __restrict__ g,
                           const float* __restrict__ b, __bf16* __restrict__ zh) {
  __shared__ float red[DMODEL];
  int n = blockIdx.x, d = threadIdx.x;
  float v = x[(size_t)n * DMODEL + d];
  red[d] = v; __syncthreads();
  for (int s = 64; s > 0; s >>= 1) { if (d < s) red[d] += red[d + s]; __syncthreads(); }
  float mu = red[0] * (1.0f / DMODEL); __syncthreads();
  float c = v - mu;
  red[d] = c * c; __syncthreads();
  for (int s = 64; s > 0; s >>= 1) { if (d < s) red[d] += red[d + s]; __syncthreads(); }
  float var = red[0] * (1.0f / DMODEL);
  zh[(size_t)n * DMODEL + d] = f2bf(c * rsqrtf(var + 1e-5f) * g[d] + b[d]);
}

// ---------------------------------------------------------------------------
// Projection: P[t] = zh @ Wk[t] + bk[t]  (bf16 out)
// Block = 64 rows x 128 cols; 8 waves -> (rowTile = w>>1, colGroup = w&1).
// A tile (16KB, contiguous) async-staged into LDS.
// ---------------------------------------------------------------------------
__global__ void __launch_bounds__(256) proj_kernel(const __bf16* __restrict__ zh,
                                                   const __bf16* __restrict__ WkT,
                                                   const float* __restrict__ bk,
                                                   __bf16* __restrict__ P) {
  __shared__ __bf16 Atile[64 * DMODEL];
  int t = blockIdx.y;
  int blockRow = blockIdx.x * 64;
  int tid = threadIdx.x;
  async_copy_tile(zh + (size_t)blockRow * DMODEL, Atile, 64 * DMODEL * 2, tid, 256);
  __syncthreads();
  int wave = tid >> 5, lane = tid & 31;
  int rowTile = wave >> 1;
  int colBase = (wave & 1) * 64;
  f32x8 acc[4] = {};
  wmma_row4<DMODEL>(Atile + rowTile * 16 * DMODEL,
                    WkT + (size_t)t * DMODEL * DMODEL + (size_t)colBase * DMODEL,
                    lane, acc);
  int khi = lane >> 4;
  __bf16* Pt = P + (size_t)t * NNODES * DMODEL;
#pragma unroll
  for (int i = 0; i < 4; ++i) {
    int col = colBase + i * 16 + (lane & 15);
    float bias = bk[t * DMODEL + col];
#pragma unroll
    for (int j = 0; j < 8; ++j) {
      int r = blockRow + rowTile * 16 + j + 8 * khi;
      if (r < NNODES) Pt[(size_t)r * DMODEL + col] = f2bf(acc[i][j] + bias);
    }
  }
}

// ---------------------------------------------------------------------------
// Edge pass A: logits + segment max (per src node, per head)
// ---------------------------------------------------------------------------
__global__ void edge_logits_kernel(const int* __restrict__ edges,
                                   const unsigned char* __restrict__ et,
                                   const __bf16* __restrict__ P,
                                   float* __restrict__ logits, float* __restrict__ m) {
  int idx = blockIdx.x * blockDim.x + threadIdx.x;
  if (idx >= NEDGES * NHEADS) return;
  int e = idx >> 3, h = idx & 7;
  int src = edges[e], dst = edges[NEDGES + e];
  int t = et[e];
  const __bf16* qp = P + ((size_t)t * NNODES + src) * DMODEL + h * 16;
  const __bf16* kp = P + ((size_t)t * NNODES + dst) * DMODEL + h * 16;
  bf16x16 q = *(const bf16x16*)qp;
  bf16x16 k = *(const bf16x16*)kp;
  float s = 0.0f;
#pragma unroll
  for (int i = 0; i < 16; ++i) s += bf2f(q[i]) * bf2f(k[i]);
  logits[idx] = s;
  atomicMaxFloat(&m[src * NHEADS + h], s);
}

// ---------------------------------------------------------------------------
// Edge pass B: scores + segment sums (denominator and numerator)
// ---------------------------------------------------------------------------
__global__ void edge_scatter_kernel(const int* __restrict__ edges,
                                    const unsigned char* __restrict__ et,
                                    const __bf16* __restrict__ P,
                                    const float* __restrict__ logits,
                                    const float* __restrict__ m,
                                    float* __restrict__ den, float* __restrict__ attn) {
  int idx = blockIdx.x * blockDim.x + threadIdx.x;
  if (idx >= NEDGES * NHEADS) return;
  int e = idx >> 3, h = idx & 7;
  int src = edges[e], dst = edges[NEDGES + e];
  int t = et[e];
  float sc = __expf(logits[idx] - m[src * NHEADS + h]);
  atomicAdd(&den[src * NHEADS + h], sc);
  const __bf16* vp = P + ((size_t)t * NNODES + dst) * DMODEL + h * 16;
  bf16x16 v = *(const bf16x16*)vp;
  float* arow = attn + (size_t)src * DMODEL + h * 16;
#pragma unroll
  for (int i = 0; i < 16; ++i) atomicAdd(&arow[i], sc * bf2f(v[i]));
}

// ---------------------------------------------------------------------------
// Finalize: x1 = x + attn/den, z2 = LN(x1) -> bf16
// ---------------------------------------------------------------------------
__global__ void finalize_kernel(const float* __restrict__ x, const float* __restrict__ attn,
                                const float* __restrict__ den, const float* __restrict__ g,
                                const float* __restrict__ b, float* __restrict__ x1,
                                __bf16* __restrict__ z2h) {
  __shared__ float red[DMODEL];
  int n = blockIdx.x, d = threadIdx.x;
  float dh = den[n * NHEADS + (d >> 4)];
  float a = attn[(size_t)n * DMODEL + d];
  float v = x[(size_t)n * DMODEL + d] + (dh > 0.0f ? a / dh : 0.0f);
  x1[(size_t)n * DMODEL + d] = v;
  red[d] = v; __syncthreads();
  for (int s = 64; s > 0; s >>= 1) { if (d < s) red[d] += red[d + s]; __syncthreads(); }
  float mu = red[0] * (1.0f / DMODEL); __syncthreads();
  float c = v - mu;
  red[d] = c * c; __syncthreads();
  for (int s = 64; s > 0; s >>= 1) { if (d < s) red[d] += red[d + s]; __syncthreads(); }
  float var = red[0] * (1.0f / DMODEL);
  z2h[(size_t)n * DMODEL + d] = f2bf(c * rsqrtf(var + 1e-5f) * g[d] + b[d]);
}

// ---------------------------------------------------------------------------
// FFN1: hidden = relu(z2 @ W1 + b1) (bf16).  Block = 64 rows x 128 cols,
// grid.y = 4 col blocks.  A tile (16KB) async-staged.
// ---------------------------------------------------------------------------
__global__ void __launch_bounds__(256) ffn1_kernel(const __bf16* __restrict__ z2h,
                                                   const __bf16* __restrict__ W1T,
                                                   const float* __restrict__ b1,
                                                   __bf16* __restrict__ hid) {
  __shared__ __bf16 Atile[64 * DMODEL];
  int blockRow = blockIdx.x * 64;
  int tid = threadIdx.x;
  async_copy_tile(z2h + (size_t)blockRow * DMODEL, Atile, 64 * DMODEL * 2, tid, 256);
  __syncthreads();
  int wave = tid >> 5, lane = tid & 31;
  int rowTile = wave >> 1;
  int colBase = blockIdx.y * 128 + (wave & 1) * 64;
  f32x8 acc[4] = {};
  wmma_row4<DMODEL>(Atile + rowTile * 16 * DMODEL,
                    W1T + (size_t)colBase * DMODEL, lane, acc);
  int khi = lane >> 4;
#pragma unroll
  for (int i = 0; i < 4; ++i) {
    int col = colBase + i * 16 + (lane & 15);
    float bias = b1[col];
#pragma unroll
    for (int j = 0; j < 8; ++j) {
      int r = blockRow + rowTile * 16 + j + 8 * khi;
      float v = acc[i][j] + bias;
      if (r < NNODES) hid[(size_t)r * NFF + col] = f2bf(v > 0.0f ? v : 0.0f);
    }
  }
}

// ---------------------------------------------------------------------------
// FFN2: out = x1 + hidden @ W2 + b2.  Block = 64 rows x 128 cols, K = 512.
// A tile (64KB) async-staged into LDS (320KB/WGP budget).
// ---------------------------------------------------------------------------
__global__ void __launch_bounds__(256) ffn2_kernel(const __bf16* __restrict__ hid,
                                                   const __bf16* __restrict__ W2T,
                                                   const float* __restrict__ b2,
                                                   const float* __restrict__ x1,
                                                   float* __restrict__ out) {
  __shared__ __bf16 Atile[64 * NFF];
  int blockRow = blockIdx.x * 64;
  int tid = threadIdx.x;
  async_copy_tile(hid + (size_t)blockRow * NFF, Atile, 64 * NFF * 2, tid, 256);
  __syncthreads();
  int wave = tid >> 5, lane = tid & 31;
  int rowTile = wave >> 1;
  int colBase = (wave & 1) * 64;
  f32x8 acc[4] = {};
  wmma_row4<NFF>(Atile + rowTile * 16 * NFF,
                 W2T + (size_t)colBase * NFF, lane, acc);
  int khi = lane >> 4;
#pragma unroll
  for (int i = 0; i < 4; ++i) {
    int col = colBase + i * 16 + (lane & 15);
    float bias = b2[col];
#pragma unroll
    for (int j = 0; j < 8; ++j) {
      int r = blockRow + rowTile * 16 + j + 8 * khi;
      if (r < NNODES)
        out[(size_t)r * DMODEL + col] = x1[(size_t)r * DMODEL + col] + acc[i][j] + bias;
    }
  }
}

// ---------------------------------------------------------------------------
extern "C" void kernel_launch(void* const* d_in, const int* in_sizes, int n_in,
                              void* d_out, int out_size, void* d_ws, size_t ws_size,
                              hipStream_t stream) {
  const float*         x      = (const float*)d_in[0];
  const int*           edges  = (const int*)d_in[1];
  const unsigned char* masks  = (const unsigned char*)d_in[2];
  const float*         Wk     = (const float*)d_in[3];
  const float*         bk     = (const float*)d_in[4];
  const float*         ln1_g  = (const float*)d_in[5];
  const float*         ln1_b  = (const float*)d_in[6];
  const float*         ln2_g  = (const float*)d_in[7];
  const float*         ln2_b  = (const float*)d_in[8];
  const float*         W1     = (const float*)d_in[9];
  const float*         b1     = (const float*)d_in[10];
  const float*         W2     = (const float*)d_in[11];
  const float*         b2     = (const float*)d_in[12];
  float*               out    = (float*)d_out;

  char* w = (char*)d_ws;
  size_t off = 0;
  auto alloc = [&](size_t bytes) -> void* {
    void* p = w + off;
    off += (bytes + 255) & ~(size_t)255;
    return p;
  };
  __bf16* zh     = (__bf16*)alloc((size_t)NNODES * DMODEL * 2);
  __bf16* WkT    = (__bf16*)alloc((size_t)NTYPES * DMODEL * DMODEL * 2);
  __bf16* W1T    = (__bf16*)alloc((size_t)NFF * DMODEL * 2);
  __bf16* W2T    = (__bf16*)alloc((size_t)DMODEL * NFF * 2);
  __bf16* P      = (__bf16*)alloc((size_t)NTYPES * NNODES * DMODEL * 2);
  float*  logits = (float*)alloc((size_t)NEDGES * NHEADS * 4);
  float*  mbuf   = (float*)alloc((size_t)NNODES * NHEADS * 4);
  float*  den    = (float*)alloc((size_t)NNODES * NHEADS * 4);
  float*  attn   = (float*)alloc((size_t)NNODES * DMODEL * 4);
  float*  x1     = (float*)alloc((size_t)NNODES * DMODEL * 4);
  __bf16* z2h    = (__bf16*)alloc((size_t)NNODES * DMODEL * 2);
  __bf16* hid    = (__bf16*)alloc((size_t)NNODES * NFF * 2);
  unsigned char* et = (unsigned char*)alloc((size_t)NEDGES);

  const int rowBlocks = (NNODES + 63) / 64;   // 782

  transpose_wk_kernel<<<(NTYPES * DMODEL * DMODEL + 255) / 256, 256, 0, stream>>>(Wk, WkT);
  transpose_w1_kernel<<<(NFF * DMODEL + 255) / 256, 256, 0, stream>>>(W1, W1T);
  transpose_w2_kernel<<<(DMODEL * NFF + 255) / 256, 256, 0, stream>>>(W2, W2T);
  etype_kernel<<<(NEDGES + 255) / 256, 256, 0, stream>>>(masks, et);
  init_kernel<<<(NNODES * DMODEL + 255) / 256, 256, 0, stream>>>(mbuf, den, attn);
  ln1_kernel<<<NNODES, DMODEL, 0, stream>>>(x, ln1_g, ln1_b, zh);
  proj_kernel<<<dim3(rowBlocks, NTYPES), 256, 0, stream>>>(zh, WkT, bk, P);
  edge_logits_kernel<<<(NEDGES * NHEADS + 255) / 256, 256, 0, stream>>>(edges, et, P, logits, mbuf);
  edge_scatter_kernel<<<(NEDGES * NHEADS + 255) / 256, 256, 0, stream>>>(edges, et, P, logits, mbuf, den, attn);
  finalize_kernel<<<NNODES, DMODEL, 0, stream>>>(x, attn, den, ln2_g, ln2_b, x1, z2h);
  ffn1_kernel<<<dim3(rowBlocks, NFF / 128), 256, 0, stream>>>(z2h, W1T, b1, hid);
  ffn2_kernel<<<rowBlocks, 256, 0, stream>>>(hid, W2T, b2, x1, out);
}
